// RewaHierarchicalAttention_90237262889101
// MI455X (gfx1250) — matrix operand order repaired and checked by reading, not compiled, not run on hardware
//
#include <hip/hip_runtime.h>
#include <hip/hip_bf16.h>
#include <math.h>

// MI455X / gfx1250, wave32. All matrix work routed through
// v_wmma_f32_16x16x32_bf16 (bf16 in, f32 accumulate).
// V staging in attention uses GLOBAL_LOAD_ASYNC_TO_LDS_B128 (ASYNCcnt-tracked).

typedef __attribute__((ext_vector_type(16))) __bf16 v16bf;
typedef __attribute__((ext_vector_type(8)))  float  v8f;
typedef __attribute__((ext_vector_type(4)))  int    v4i;

#define B_   4
#define N_   4096
#define E_   1024
#define H_   16
#define D_   64
#define BH_  64          // B_*H_
#define M_   16384       // B_*N_

#if __has_builtin(__builtin_amdgcn_global_load_async_to_lds_b128)
#define HAVE_ASYNC_LDS 1
#else
#define HAVE_ASYNC_LDS 0
#endif

static __device__ inline void wait_asynccnt0() {
#if __has_builtin(__builtin_amdgcn_s_wait_asynccnt)
  __builtin_amdgcn_s_wait_asynccnt(0);
#else
  asm volatile("s_wait_asynccnt 0x0" ::: "memory");
#endif
}

union Frag {
  v16bf v;
  uint4 q[2];
  __bf16 e[16];
};

union Row64 {
  uint4 q[4];
  __bf16 e[64];
};

static __device__ inline v8f wmma_bf16(v16bf a, v16bf b, v8f c) {
  // D = A(16x32 bf16) * B(32x16 bf16) + C(16x16 f32)
  return __builtin_amdgcn_wmma_f32_16x16x32_bf16(false, a, false, b,
                                                 (short)0, c, false, false);
}

// A fragment: lane l -> row (l&15); 16 contiguous K values at kbase+hi*16.
static __device__ inline Frag load_a(const __bf16* __restrict__ rowptr,
                                     int kbase, int hi) {
  Frag f;
  f.q[0] = *(const uint4*)(rowptr + kbase + hi * 16);
  f.q[1] = *(const uint4*)(rowptr + kbase + hi * 16 + 8);
  return f;
}

// B fragment from column-major storage: lane l holds column (l&15),
// 16 contiguous K values starting at kbase + hi*16 (32 bytes).
static __device__ inline Frag load_b(const __bf16* __restrict__ colptr,
                                     int kbase, int hi) {
  Frag f;
  const uint4* p = (const uint4*)(colptr + kbase + hi * 16);
  f.q[0] = p[0];
  f.q[1] = p[1];
  return f;
}

// ---------------------------------------------------------------- converts
__global__ void cvt_f32_bf16_kernel(const float* __restrict__ in,
                                    __bf16* __restrict__ out, int n) {
  int i = blockIdx.x * blockDim.x + threadIdx.x;
  if (i < n) out[i] = (__bf16)in[i];
}

// W[k][c] (f32) -> WT[c][k] (bf16), square E_xE_
__global__ void cvt_w_t_kernel(const float* __restrict__ W,
                               __bf16* __restrict__ WT) {
  int i = blockIdx.x * blockDim.x + threadIdx.x;
  if (i >= E_ * E_) return;
  int c = i >> 10;        // output column (row of WT)
  int k = i & (E_ - 1);
  WT[i] = (__bf16)W[k * E_ + c];
}

// probes[lvl][h][d][p] (f32) -> probeT[lvl][h][p][d] (bf16)
__global__ void cvt_probe_t_kernel(const float* __restrict__ P,
                                   __bf16* __restrict__ PT) {
  int i = blockIdx.x * blockDim.x + threadIdx.x;
  if (i >= 3 * 16 * 64 * 64) return;
  int d   = i & 63;
  int p   = (i >> 6) & 63;
  int hh  = (i >> 12) & 15;
  int lvl = i >> 16;
  PT[i] = (__bf16)P[(((size_t)lvl * 16 + hh) * 64 + d) * 64 + p];
}

__global__ void zero_f32_kernel(float* __restrict__ p, int n) {
  int i = blockIdx.x * blockDim.x + threadIdx.x;
  if (i < n) p[i] = 0.f;
}

// --------------------------------------------------------------- main GEMM
// One wave computes a 32x64 output tile of A[M_,E_] @ WT^T + bias
// (two A-row streams share the four B streams -> halved B traffic).
// mode 0: write bf16 in head layout [bh][n][d]; mode 1: write f32 flat.
__global__ __launch_bounds__(32) void gemm_wmma_kernel(
    const __bf16* __restrict__ A, const __bf16* __restrict__ WT,
    const float* __restrict__ bias, void* __restrict__ outp, int mode) {
  int lane = threadIdx.x;
  int l15 = lane & 15, hi = lane >> 4;
  int colbase = blockIdx.x * 64;
  int mbase = blockIdx.y * 32;
  const __bf16* arow0 = A + (size_t)(mbase + l15) * E_;
  const __bf16* arow1 = A + (size_t)(mbase + 16 + l15) * E_;
  const __bf16* w0 = WT + (size_t)(colbase + 0 + l15) * E_;
  const __bf16* w1 = WT + (size_t)(colbase + 16 + l15) * E_;
  const __bf16* w2 = WT + (size_t)(colbase + 32 + l15) * E_;
  const __bf16* w3 = WT + (size_t)(colbase + 48 + l15) * E_;
  v8f c00 = {}, c01 = {}, c02 = {}, c03 = {};
  v8f c10 = {}, c11 = {}, c12 = {}, c13 = {};
  for (int kk = 0; kk < E_; kk += 32) {
    if (kk + 256 < E_) {               // pull next K-blocks toward the WGP
      __builtin_prefetch(arow0 + kk + 256, 0, 1);
      __builtin_prefetch(arow1 + kk + 256, 0, 1);
    }
    Frag a0 = load_a(arow0, kk, hi);
    Frag a1 = load_a(arow1, kk, hi);
    Frag b0 = load_b(w0, kk, hi);
    Frag b1 = load_b(w1, kk, hi);
    Frag b2 = load_b(w2, kk, hi);
    Frag b3 = load_b(w3, kk, hi);
    c00 = wmma_bf16(a0.v, b0.v, c00);
    c01 = wmma_bf16(a0.v, b1.v, c01);
    c02 = wmma_bf16(a0.v, b2.v, c02);
    c03 = wmma_bf16(a0.v, b3.v, c03);
    c10 = wmma_bf16(a1.v, b0.v, c10);
    c11 = wmma_bf16(a1.v, b1.v, c11);
    c12 = wmma_bf16(a1.v, b2.v, c12);
    c13 = wmma_bf16(a1.v, b3.v, c13);
  }
  // epilogue: C layout -> (row = j + hi*8, col = tile*16 + l15)
  for (int half = 0; half < 2; ++half) {
    for (int j = 0; j < 8; ++j) {
      int m = mbase + half * 16 + j + hi * 8;
      int b = m >> 12, n = m & (N_ - 1);
      float v[4];
      if (half == 0) { v[0] = c00[j]; v[1] = c01[j]; v[2] = c02[j]; v[3] = c03[j]; }
      else           { v[0] = c10[j]; v[1] = c11[j]; v[2] = c12[j]; v[3] = c13[j]; }
      for (int t = 0; t < 4; ++t) {
        int col = colbase + t * 16 + l15;
        float val = v[t] + bias[col];
        if (mode == 0) {
          int h = col >> 6, d = col & 63;
          ((__bf16*)outp)[(((size_t)(b * 16 + h)) * N_ + n) * D_ + d] = (__bf16)val;
        } else {
          ((float*)outp)[(size_t)m * E_ + col] = val;
        }
      }
    }
  }
}

// ------------------------------------------------------- probe -> bucket id
__global__ __launch_bounds__(32) void proj_bucket_kernel(
    const __bf16* __restrict__ qh, const __bf16* __restrict__ pT,
    int* __restrict__ bucket, int nb) {
  __shared__ float Cb[16 * 64];
  int lane = threadIdx.x, l15 = lane & 15, hi = lane >> 4;
  int bh = blockIdx.y;
  int rowbase = blockIdx.x * 16;
  int hh = bh & 15;
  const __bf16* qrow = qh + ((size_t)bh * N_ + rowbase + l15) * D_;
  const __bf16* pb = pT + (size_t)hh * 64 * 64;
  v8f c0 = {}, c1 = {}, c2 = {}, c3 = {};
  for (int s = 0; s < 2; ++s) {
    Frag a = load_a(qrow, s * 32, hi);
    Frag b0 = load_b(pb + (0 + l15) * 64, s * 32, hi);
    Frag b1 = load_b(pb + (16 + l15) * 64, s * 32, hi);
    Frag b2 = load_b(pb + (32 + l15) * 64, s * 32, hi);
    Frag b3 = load_b(pb + (48 + l15) * 64, s * 32, hi);
    c0 = wmma_bf16(a.v, b0.v, c0);
    c1 = wmma_bf16(a.v, b1.v, c1);
    c2 = wmma_bf16(a.v, b2.v, c2);
    c3 = wmma_bf16(a.v, b3.v, c3);
  }
  for (int j = 0; j < 8; ++j) {
    int r = j + hi * 8;
    Cb[r * 64 + 0 + l15]  = c0[j];
    Cb[r * 64 + 16 + l15] = c1[j];
    Cb[r * 64 + 32 + l15] = c2[j];
    Cb[r * 64 + 48 + l15] = c3[j];
  }
  __syncthreads();
  if (hi == 0) {
    float best = -3.0e38f;
    int arg = 0;
    for (int p = 0; p < 64; ++p) {
      float v = Cb[l15 * 64 + p];
      if (v > best) { best = v; arg = p; }
    }
    bucket[(size_t)bh * N_ + rowbase + l15] = (arg * 9973) % nb;
  }
}

// -------------------------------------------------- stable counting sort
__global__ __launch_bounds__(64) void bucket_sort_kernel(
    const int* __restrict__ bucket, int* __restrict__ sidx, int nb) {
  __shared__ unsigned short cnt[64 * 256];
  __shared__ int start[256];
  int t = threadIdx.x;
  int bh = blockIdx.x;
  const int* bk = bucket + (size_t)bh * N_;
  for (int i = t; i < 64 * nb; i += 64) cnt[i] = 0;
  __syncthreads();
  for (int i = 0; i < 64; ++i) {
    int b = bk[t * 64 + i];
    cnt[t * nb + b]++;
  }
  __syncthreads();
  for (int b = t; b < nb; b += 64) {
    int s = 0;
    for (int tt = 0; tt < 64; ++tt) s += cnt[tt * nb + b];
    start[b] = s;
  }
  __syncthreads();
  if (t == 0) {
    int run = 0;
    for (int b = 0; b < nb; ++b) { int s = start[b]; start[b] = run; run += s; }
  }
  __syncthreads();
  for (int b = t; b < nb; b += 64) {
    int run = start[b];
    for (int tt = 0; tt < 64; ++tt) {
      int c = cnt[tt * nb + b];
      cnt[tt * nb + b] = (unsigned short)run;
      run += c;
    }
  }
  __syncthreads();
  for (int i = 0; i < 64; ++i) {
    int n = t * 64 + i;
    int b = bk[n];
    int pos = cnt[t * nb + b];
    cnt[t * nb + b] = (unsigned short)(pos + 1);
    sidx[(size_t)bh * N_ + pos] = n;
  }
}

// ------------------------------------------- flash-style local attention
// One wave handles 16 sorted query rows; streams keys in 32-key blocks.
__global__ __launch_bounds__(32) void attn_kernel(
    const __bf16* __restrict__ qh, const __bf16* __restrict__ kh,
    const __bf16* __restrict__ vh, const int* __restrict__ sidx,
    float* __restrict__ acc, int cs, int nc) {
  __shared__ __align__(16) float  Sb[16 * 32];
  __shared__ __align__(16) __bf16 Pb[16 * 32];
  __shared__ __align__(16) __bf16 Vbuf[32 * 64];  // async: [key][d]; fallback: [d][key]
  __shared__ float mrow[16], lrow[16], fsc[16];
  __shared__ int nbuf[16], kn[32];

  int lane = threadIdx.x, l15 = lane & 15, hi = lane >> 4;
  int bh = blockIdx.y;
  int rowbase = blockIdx.x * 16;
  int chunk = rowbase / cs;
  const int* sid = sidx + (size_t)bh * N_;

  int qn = sid[rowbase + l15];
  if (hi == 0) { nbuf[l15] = qn; mrow[l15] = -3.0e38f; lrow[l15] = 0.f; }
  const __bf16* qrow = qh + ((size_t)bh * N_ + qn) * D_;
  Frag aq0 = load_a(qrow, 0, hi);
  Frag aq1 = load_a(qrow, 32, hi);
  v8f o0 = {}, o1 = {}, o2 = {}, o3 = {};
  __syncthreads();

  int K2 = 2 * cs;
  for (int kb = 0; kb < K2; kb += 32) {
    if (chunk == 0 && kb + 32 <= cs) continue;  // fully-masked prev block
    // gather sorted key ids for this 32-key block
    {
      int j = kb + lane;
      int pos;
      if (j < cs) pos = ((chunk == 0 ? nc - 1 : chunk - 1) * cs) + j;
      else        pos = chunk * cs + (j - cs);
      kn[lane] = sid[pos];
    }
    __syncthreads();

#if HAVE_ASYNC_LDS
    // kick off async V-row gather: global (per-lane row) -> LDS [key][d]
    {
      const __bf16* vrow = vh + ((size_t)bh * N_ + kn[lane]) * D_;
      v4i __attribute__((address_space(1)))* gsrc =
          (v4i __attribute__((address_space(1)))*)vrow;
      v4i __attribute__((address_space(3)))* ldst =
          (v4i __attribute__((address_space(3)))*)&Vbuf[lane * 64];
      __builtin_amdgcn_global_load_async_to_lds_b128(gsrc, ldst, 0, 0);
      __builtin_amdgcn_global_load_async_to_lds_b128(gsrc, ldst, 16, 0);
      __builtin_amdgcn_global_load_async_to_lds_b128(gsrc, ldst, 32, 0);
      __builtin_amdgcn_global_load_async_to_lds_b128(gsrc, ldst, 48, 0);
    }
#endif

    // scores S = Q K^T (two 16x16 tiles, two d-steps each)
    v8f s0 = {}, s1 = {};
    {
      const __bf16* k0 = kh + ((size_t)bh * N_ + kn[l15]) * D_;
      const __bf16* k1 = kh + ((size_t)bh * N_ + kn[16 + l15]) * D_;
      Frag b;
      b = load_b(k0, 0, hi);  s0 = wmma_bf16(aq0.v, b.v, s0);
      b = load_b(k0, 32, hi); s0 = wmma_bf16(aq1.v, b.v, s0);
      b = load_b(k1, 0, hi);  s1 = wmma_bf16(aq0.v, b.v, s1);
      b = load_b(k1, 32, hi); s1 = wmma_bf16(aq1.v, b.v, s1);
    }
    for (int j = 0; j < 8; ++j) {
      Sb[(j + hi * 8) * 32 + l15]      = s0[j] * 0.125f;
      Sb[(j + hi * 8) * 32 + 16 + l15] = s1[j] * 0.125f;
    }
    __syncthreads();

    // online softmax (lanes 0..15, one row each)
    if (hi == 0) {
      int r = l15;
      float mold = mrow[r], mnew = mold;
      for (int c = 0; c < 32; ++c) {
        bool msk = (chunk == 0) && (kb + c < cs);
        if (!msk) mnew = fmaxf(mnew, Sb[r * 32 + c]);
      }
      float sum = 0.f;
      for (int c = 0; c < 32; ++c) {
        bool msk = (chunk == 0) && (kb + c < cs);
        float p = msk ? 0.f : __expf(Sb[r * 32 + c] - mnew);
        Pb[r * 32 + c] = (__bf16)p;
        sum += p;
      }
      float f = __expf(mold - mnew);
      fsc[r] = f;
      lrow[r] = lrow[r] * f + sum;
      mrow[r] = mnew;
    }
#if !HAVE_ASYNC_LDS
    // fallback: VGPR gather + transposed LDS stage ([d][key])
    {
      Row64 vr;
      const __bf16* vrow = vh + ((size_t)bh * N_ + kn[lane]) * D_;
      const uint4* vp = (const uint4*)vrow;
      vr.q[0] = vp[0]; vr.q[1] = vp[1]; vr.q[2] = vp[2]; vr.q[3] = vp[3];
      for (int d = 0; d < 64; ++d) Vbuf[d * 32 + lane] = vr.e[d];
    }
#else
    wait_asynccnt0();
#endif
    __syncthreads();

    // rescale running output by exp(mold-mnew)
    float fr[8];
    for (int j = 0; j < 8; ++j) fr[j] = fsc[j + hi * 8];
    for (int j = 0; j < 8; ++j) {
      o0[j] *= fr[j]; o1[j] *= fr[j]; o2[j] *= fr[j]; o3[j] *= fr[j];
    }
    // P fragment (A: 16 rows x 32 keys) from LDS
    Frag pa;
    pa.q[0] = *(const uint4*)&Pb[l15 * 32 + hi * 16];
    pa.q[1] = *(const uint4*)&Pb[l15 * 32 + hi * 16 + 8];
    // O += P * V  (4 d-tiles)
    for (int t = 0; t < 4; ++t) {
      Frag b;
#if HAVE_ASYNC_LDS
      // Vbuf is [key][d] row-major: lane -> column d = t*16+l15, 16 keys
      for (int kk2 = 0; kk2 < 16; ++kk2)
        b.e[kk2] = Vbuf[(hi * 16 + kk2) * 64 + t * 16 + l15];
#else
      // Vbuf is [d][key]: 32B contiguous per lane
      const uint4* vp = (const uint4*)&Vbuf[(t * 16 + l15) * 32 + hi * 16];
      b.q[0] = vp[0];
      b.q[1] = vp[1];
#endif
      if (t == 0) o0 = wmma_bf16(pa.v, b.v, o0);
      if (t == 1) o1 = wmma_bf16(pa.v, b.v, o1);
      if (t == 2) o2 = wmma_bf16(pa.v, b.v, o2);
      if (t == 3) o3 = wmma_bf16(pa.v, b.v, o3);
    }
    __syncthreads();
  }

  // normalize and scatter-accumulate (unsort) into acc[bh][n][d]
  for (int j = 0; j < 8; ++j) {
    int r = j + hi * 8;
    float inv = 1.f / lrow[r];
    int n = nbuf[r];
    float* ap = acc + ((size_t)bh * N_ + n) * D_ + l15;
    ap[0]  += o0[j] * inv;
    ap[16] += o1[j] * inv;
    ap[32] += o2[j] * inv;
    ap[48] += o3[j] * inv;
  }
}

// merge heads: ctx[b][n][h*64+d] = acc[bh][n][d] / 3  (bf16)
__global__ void merge_ctx_kernel(const float* __restrict__ acc,
                                 __bf16* __restrict__ ctx) {
  int i = blockIdx.x * blockDim.x + threadIdx.x;
  if (i >= M_ * E_) return;
  int d = i & 63;
  int h = (i >> 6) & 15;
  int n = (i >> 10) & (N_ - 1);
  int b = i >> 22;
  ctx[i] = (__bf16)(acc[(((size_t)(b * 16 + h)) * N_ + n) * D_ + d] * (1.f / 3.f));
}

// ------------------------------------------------------------------ launch
extern "C" void kernel_launch(void* const* d_in, const int* in_sizes, int n_in,
                              void* d_out, int out_size, void* d_ws, size_t ws_size,
                              hipStream_t stream) {
  const float* x  = (const float*)d_in[0];
  const float* Wq = (const float*)d_in[1];
  const float* bq = (const float*)d_in[2];
  const float* Wk = (const float*)d_in[3];
  const float* bk = (const float*)d_in[4];
  const float* Wv = (const float*)d_in[5];
  const float* bv = (const float*)d_in[6];
  const float* Wo = (const float*)d_in[7];
  const float* bo = (const float*)d_in[8];
  const float* probes = (const float*)d_in[9];

  char* ws = (char*)d_ws;
  size_t off = 0;
  auto alloc = [&](size_t bytes) {
    void* p = ws + off;
    off = (off + bytes + 255) & ~(size_t)255;
    return p;
  };
  __bf16* xb  = (__bf16*)alloc((size_t)M_ * E_ * 2);
  __bf16* WqT = (__bf16*)alloc((size_t)E_ * E_ * 2);
  __bf16* WkT = (__bf16*)alloc((size_t)E_ * E_ * 2);
  __bf16* WvT = (__bf16*)alloc((size_t)E_ * E_ * 2);
  __bf16* WoT = (__bf16*)alloc((size_t)E_ * E_ * 2);
  __bf16* pT  = (__bf16*)alloc((size_t)3 * 16 * 64 * 64 * 2);
  __bf16* qh  = (__bf16*)alloc((size_t)BH_ * N_ * D_ * 2);
  __bf16* kh  = (__bf16*)alloc((size_t)BH_ * N_ * D_ * 2);
  __bf16* vh  = (__bf16*)alloc((size_t)BH_ * N_ * D_ * 2);
  float*  acc = (float*)alloc((size_t)BH_ * N_ * D_ * 4);
  int*    bkt = (int*)alloc((size_t)BH_ * N_ * 4);
  int*    sdx = (int*)alloc((size_t)BH_ * N_ * 4);
  __bf16* ctx = (__bf16*)alloc((size_t)M_ * E_ * 2);

  // converts
  {
    int n = M_ * E_;
    cvt_f32_bf16_kernel<<<(n + 255) / 256, 256, 0, stream>>>(x, xb, n);
    int nw = E_ * E_;
    cvt_w_t_kernel<<<(nw + 255) / 256, 256, 0, stream>>>(Wq, WqT);
    cvt_w_t_kernel<<<(nw + 255) / 256, 256, 0, stream>>>(Wk, WkT);
    cvt_w_t_kernel<<<(nw + 255) / 256, 256, 0, stream>>>(Wv, WvT);
    cvt_w_t_kernel<<<(nw + 255) / 256, 256, 0, stream>>>(Wo, WoT);
    int np = 3 * 16 * 64 * 64;
    cvt_probe_t_kernel<<<(np + 255) / 256, 256, 0, stream>>>(probes, pT);
    int na = BH_ * N_ * D_;
    zero_f32_kernel<<<(na + 255) / 256, 256, 0, stream>>>(acc, na);
  }

  // QKV projections (head-layout bf16 outputs)
  dim3 ggrid(E_ / 64, M_ / 32);
  gemm_wmma_kernel<<<ggrid, 32, 0, stream>>>(xb, WqT, bq, qh, 0);
  gemm_wmma_kernel<<<ggrid, 32, 0, stream>>>(xb, WkT, bk, kh, 0);
  gemm_wmma_kernel<<<ggrid, 32, 0, stream>>>(xb, WvT, bv, vh, 0);

  // per-level: probe projection -> bucket, stable sort, local attention
  const int css[3] = {256, 64, 16};
  for (int lvl = 0; lvl < 3; ++lvl) {
    int cs = css[lvl];
    int nb = N_ / cs;
    int nc = N_ / cs;
    proj_bucket_kernel<<<dim3(N_ / 16, BH_), 32, 0, stream>>>(
        qh, pT + (size_t)lvl * 16 * 64 * 64, bkt, nb);
    bucket_sort_kernel<<<BH_, 64, 0, stream>>>(bkt, sdx, nb);
    attn_kernel<<<dim3(N_ / 16, BH_), 32, 0, stream>>>(qh, kh, vh, sdx, acc,
                                                       cs, nc);
  }

  // merge heads + output projection (f32 to d_out)
  {
    int n = M_ * E_;
    merge_ctx_kernel<<<(n + 255) / 256, 256, 0, stream>>>(acc, ctx);
    gemm_wmma_kernel<<<ggrid, 32, 0, stream>>>(ctx, WoT, bo, d_out, 1);
  }
}
